// FAVOR_91319594647802
// MI455X (gfx1250) — compile-verified, matrix operand order checked by reading, not run on hardware
//
#include <hip/hip_runtime.h>

// ---------------- problem constants (fixed by the reference) ----------------
constexpr int Bn  = 8;              // batch
constexpr int Dn  = 64;             // feature dim
constexpr int Ln  = 2048;           // sequence length
constexpr int Mn  = 64;             // projection features
constexpr int Vn  = 64;             // value dim
constexpr int Cn  = 64;             // chunk (time tile) length
constexpr int NCH = Ln / Cn;        // 32 sequential chunks
constexpr int VSL = 16;             // V columns per block
constexpr int NSL = Vn / VSL;       // 4 V-slices  -> 32 blocks total
constexpr int LDB = 72;             // bf16 LDS row stride (padded)
constexpr int LDV = 68;             // f32 V-slice LDS stride (padded, = TDM pad)
constexpr int LDS_S = 17;           // f32 state row stride (padded)

typedef __attribute__((ext_vector_type(16))) __bf16         v16bf;
typedef __attribute__((ext_vector_type(16))) unsigned short v16us;
typedef __attribute__((ext_vector_type(8)))  float          v8f;
typedef __attribute__((ext_vector_type(4)))  unsigned int   v4u;
typedef __attribute__((ext_vector_type(8)))  int            v8i;
typedef __attribute__((ext_vector_type(4)))  int            v4i;

#if __has_builtin(__builtin_amdgcn_tensor_load_to_lds)
#define USE_TDM 1
#else
#define USE_TDM 0
#endif

union BFV { v16us u; v16bf b; };

__device__ __forceinline__ unsigned short f2bf(float x) {
  __bf16 h = (__bf16)x;                       // native v_cvt, RNE
  return __builtin_bit_cast(unsigned short, h);
}
__device__ __forceinline__ float bf2f(unsigned short h) {
  return __uint_as_float(((unsigned)h) << 16);
}

#define WMMA_BF16(A_, B_, Acc_) \
  __builtin_amdgcn_wmma_f32_16x16x32_bf16(false, (A_), false, (B_), (short)0, (Acc_), false, false)

// ---- fragment gathers per CDNA5 ISA 16-bit A/B layouts ----------
// A operand (16x32, row-striped): lane(0..15 | 16..31) holds row = tile*16 + (lane&15),
// VGPR j holds K pair k = (j>>2)*16 + (lane>>4)*8 + (j&3)*2.
// B operand (32x16) uses the mirrored layout with N in place of M, so a source
// stored N-major [n][k] uses the same gather.
__device__ __forceinline__ v16bf frag_rowmajor_us(const unsigned short* src, int ld,
                                                  int tile, int kbase, int lane) {
  const int half = lane >> 4, rr = lane & 15;
  const unsigned short* p = src + (tile * 16 + rr) * ld + kbase;
  BFV t;
#pragma unroll
  for (int j = 0; j < 8; ++j) {
    const int k = ((j >> 2) << 4) + (half << 3) + ((j & 3) << 1);
    t.u[2 * j]     = p[k];
    t.u[2 * j + 1] = p[k + 1];
  }
  return t.b;
}
// A operand whose logical matrix is the transpose of the stored one: elem = src[k][row]
__device__ __forceinline__ v16bf frag_kmajor_us(const unsigned short* src, int ld,
                                                int tile, int kbase, int lane) {
  const int half = lane >> 4, rr = lane & 15;
  const unsigned short* p = src + tile * 16 + rr;
  BFV t;
#pragma unroll
  for (int j = 0; j < 8; ++j) {
    const int k = kbase + ((j >> 2) << 4) + (half << 3) + ((j & 3) << 1);
    t.u[2 * j]     = p[k * ld];
    t.u[2 * j + 1] = p[(k + 1) * ld];
  }
  return t.b;
}
// B operand from an N-major f32 source: elem = src[n][k]
__device__ __forceinline__ v16bf frag_nmajor_f32(const float* src, int ld,
                                                 int tile, int kbase, int lane) {
  const int half = lane >> 4, nn = lane & 15;
  const float* p = src + (tile * 16 + nn) * ld + kbase;
  BFV t;
#pragma unroll
  for (int j = 0; j < 8; ++j) {
    const int k = ((j >> 2) << 4) + (half << 3) + ((j & 3) << 1);
    t.u[2 * j]     = f2bf(p[k]);
    t.u[2 * j + 1] = f2bf(p[k + 1]);
  }
  return t.b;
}
// B operand from a K-major f32 source: elem = src[k][n]
__device__ __forceinline__ v16bf frag_kmajor_f32(const float* src, int ld,
                                                 int tile, int kbase, int lane) {
  const int half = lane >> 4, nn = lane & 15;
  const float* p = src + tile * 16 + nn;
  BFV t;
#pragma unroll
  for (int j = 0; j < 8; ++j) {
    const int k = kbase + ((j >> 2) << 4) + (half << 3) + ((j & 3) << 1);
    t.u[2 * j]     = f2bf(p[k * ld]);
    t.u[2 * j + 1] = f2bf(p[(k + 1) * ld]);
  }
  return t.b;
}

// ---------------------------------------------------------------------------
// One block = (batch b, V-slice vs). 256 threads = 8 wave32s.
// Sequential scan over 32 chunks; recurrent state S (64x16 f32) + z (64 f32) in LDS.
// ---------------------------------------------------------------------------
__global__ __launch_bounds__(256)
void favor_causal_kernel(const float* __restrict__ keys,
                         const float* __restrict__ values,
                         const float* __restrict__ queries,
                         const float* __restrict__ proj,
                         float* __restrict__ out) {
  __shared__ unsigned short sQT[Cn * LDB];   // bf16 bits: Q chunk, [i][d]
  __shared__ unsigned short sKT[Cn * LDB];   // bf16 bits: K chunk, [i][d]
  __shared__ unsigned short sPQ[Cn * LDB];   // bf16 bits: phi_q,   [i][m]
  __shared__ unsigned short sPK[Cn * LDB];   // bf16 bits: phi_k,   [i][m]
  __shared__ unsigned short sA [Cn * LDB];   // bf16 bits: masked A,[l][i]
  __shared__ float sV[VSL * LDV];            // f32: V slice, [v][i]  (TDM-padded rows)
  __shared__ float sS[Mn * LDS_S];           // f32 state:    [m][v]
  __shared__ float sZ[Mn];                   // f32 state:    z[m]
  __shared__ float sNorm[Cn];                // 1/denominator per row

  const int tid  = threadIdx.x;
  const int lane = tid & 31;
  const int w    = tid >> 5;                 // wave 0..7
  const int half = lane >> 4;
  const int n    = lane & 15;
  const int b    = blockIdx.x >> 2;
  const int vs   = blockIdx.x & 3;

  const float* kb = keys    + (size_t)b * Dn * Ln;
  const float* qb = queries + (size_t)b * Dn * Ln;
  const float* vb = values  + (size_t)b * Vn * Ln + (size_t)vs * VSL * Ln;
  float*       ob = out     + (size_t)b * Vn * Ln + (size_t)vs * VSL * Ln;

  // zero recurrent state
  for (int i = tid; i < Mn * LDS_S; i += 256) sS[i] = 0.f;
  if (tid < Mn) sZ[tid] = 0.f;

  // Persistent projection-matrix B-fragments (K-major f32 source W[d][m], ld=Mn).
  // Wave w owns output tiles (r, c0) and (r, c0+1) in all 64-wide GEMMs.
  const int r  = w >> 1;
  const int c0 = (w & 1) << 1;
  const v16bf wf00 = frag_kmajor_f32(proj, Mn, c0,     0,  lane);
  const v16bf wf01 = frag_kmajor_f32(proj, Mn, c0,     32, lane);
  const v16bf wf10 = frag_kmajor_f32(proj, Mn, c0 + 1, 0,  lane);
  const v16bf wf11 = frag_kmajor_f32(proj, Mn, c0 + 1, 32, lane);

  __syncthreads();

  for (int ch = 0; ch < NCH; ++ch) {
    const int l0 = ch * Cn;

    // ---- phase 0a: TDM — async DMA the 16x64 f32 V tile into padded LDS ----
#if USE_TDM
    if (w == 0) {
      const unsigned lds_v = (unsigned)(size_t)(&sV[0]);   // LDS byte offset (low 32b)
      const unsigned long long ga = (unsigned long long)(size_t)(vb + l0);
      v4u g0;
      g0[0] = 1u;                                          // count=1, user descriptor
      g0[1] = lds_v;                                       // lds_addr
      g0[2] = (unsigned)(ga & 0xFFFFFFFFu);                // global_addr[31:0]
      g0[3] = (unsigned)((ga >> 32) & 0x01FFFFFFu)         // global_addr[56:32]
            | (2u << 30);                                  // type = 2 ("image")
      v8i g1;
      g1[0] = (int)((2u << 16)      // data_size = 4B
                  | (1u << 20)      // pad_enable
                  | (5u << 22)      // pad_interval: 64 DWORDs (one 64-elem row)
                  | (3u << 25));    // pad_amount:   4 DWORDs  -> LDS stride 68 f32
      g1[1] = (int)((unsigned)Ln << 16);   // tensor_dim0[15:0] = 2048
      g1[2] = (int)((unsigned)VSL << 16);  // tensor_dim0 hi = 0 | tensor_dim1 = 16
      g1[3] = (int)((unsigned)Cn << 16);   // tensor_dim1 hi = 0 | tile_dim0 = 64
      g1[4] = VSL;                         // tile_dim1 = 16, tile_dim2 = 0
      g1[5] = Ln;                          // tensor_dim0_stride = 2048
      g1[6] = 0;                           // stride hi | tensor_dim1_stride lo
      g1[7] = 0;
      const v4i gz4 = {0, 0, 0, 0};
      const v8i gz8 = {0, 0, 0, 0, 0, 0, 0, 0};
      // 6-arg toolchain form: (g0, g1, g2, g3, g4, cpol)
      __builtin_amdgcn_tensor_load_to_lds(g0, g1, gz4, gz4, gz8, 0);
    }
#else
    {
      const int vr = tid >> 4, i4 = (tid & 15) << 2;
      *(float4*)&sV[vr * LDV + i4] = *(const float4*)(vb + vr * Ln + l0 + i4);
    }
#endif

    // ---- phase 0b: stage Q/K chunk (global f32 -> LDS bf16, transposed) ----
#pragma unroll
    for (int it = 0; it < 4; ++it) {
      const int idx = tid + (it << 8);            // 0..1023
      const int d   = idx >> 4;
      const int i4  = (idx & 15) << 2;
      const float4 qv = *(const float4*)(qb + d * Ln + l0 + i4);
      const float4 kv = *(const float4*)(kb + d * Ln + l0 + i4);
      sQT[(i4 + 0) * LDB + d] = f2bf(qv.x);
      sQT[(i4 + 1) * LDB + d] = f2bf(qv.y);
      sQT[(i4 + 2) * LDB + d] = f2bf(qv.z);
      sQT[(i4 + 3) * LDB + d] = f2bf(qv.w);
      sKT[(i4 + 0) * LDB + d] = f2bf(kv.x);
      sKT[(i4 + 1) * LDB + d] = f2bf(kv.y);
      sKT[(i4 + 2) * LDB + d] = f2bf(kv.z);
      sKT[(i4 + 3) * LDB + d] = f2bf(kv.w);
    }
    if (ch + 1 < NCH) {                           // pull next chunk toward L2/L0
      const int d = tid >> 2, i4 = (tid & 3) << 4;
      __builtin_prefetch(qb + d * Ln + l0 + Cn + i4, 0, 1);
      __builtin_prefetch(kb + d * Ln + l0 + Cn + i4, 0, 1);
    }
#if USE_TDM
    if (w == 0) __builtin_amdgcn_s_wait_tensorcnt(0);  // own TENSORcnt, pre-barrier
#endif
    __syncthreads();

    // ---- phase 1: phi = relu(X^T W) + eps, for Q and K (each wave: 2 tiles) ----
    {
      const v16bf aq0 = frag_rowmajor_us(sQT, LDB, r, 0,  lane);
      const v16bf aq1 = frag_rowmajor_us(sQT, LDB, r, 32, lane);
      const v16bf ak0 = frag_rowmajor_us(sKT, LDB, r, 0,  lane);
      const v16bf ak1 = frag_rowmajor_us(sKT, LDB, r, 32, lane);
#pragma unroll
      for (int cc = 0; cc < 2; ++cc) {
        const v16bf w0 = cc ? wf10 : wf00;
        const v16bf w1 = cc ? wf11 : wf01;
        v8f accq = {};
        accq = WMMA_BF16(aq0, w0, accq);
        accq = WMMA_BF16(aq1, w1, accq);
        v8f acck = {};
        acck = WMMA_BF16(ak0, w0, acck);
        acck = WMMA_BF16(ak1, w1, acck);
#pragma unroll
        for (int j = 0; j < 8; ++j) {
          const int row = r * 16 + (half << 3) + j;
          const int col = (c0 + cc) * 16 + n;
          sPQ[row * LDB + col] = f2bf(fmaxf(accq[j], 0.f) + 1.0e-3f);
          sPK[row * LDB + col] = f2bf(fmaxf(acck[j], 0.f) + 1.0e-3f);
        }
      }
    }
    __syncthreads();

    // ---- phase 2: A = phi_q phi_k^T, causal-masked, -> sA (bf16) ----
    {
      const v16bf a0 = frag_rowmajor_us(sPQ, LDB, r, 0,  lane);
      const v16bf a1 = frag_rowmajor_us(sPQ, LDB, r, 32, lane);
#pragma unroll
      for (int cc = 0; cc < 2; ++cc) {
        const int ct = c0 + cc;
        const v16bf b0 = frag_rowmajor_us(sPK, LDB, ct, 0,  lane);  // B[m][i]=phi_k[i][m]
        const v16bf b1 = frag_rowmajor_us(sPK, LDB, ct, 32, lane);
        v8f acc = {};
        acc = WMMA_BF16(a0, b0, acc);
        acc = WMMA_BF16(a1, b1, acc);
#pragma unroll
        for (int j = 0; j < 8; ++j) {
          const int row = r * 16 + (half << 3) + j;
          const int col = ct * 16 + n;
          sA[row * LDB + col] = f2bf((col <= row) ? acc[j] : 0.f);
        }
      }
    }
    __syncthreads();

    // ---- phase 3: norms (rows) and z column-sums, scalar side channel ----
    float zacc = 0.f;
    if (tid < Cn) {
      float s = 0.f;
      for (int m = 0; m < Mn; ++m) s += bf2f(sPQ[tid * LDB + m]) * sZ[m];
      for (int i = 0; i < Cn; ++i) s += bf2f(sA[tid * LDB + i]);
      sNorm[tid] = 1.0f / s;                     // EPS==0 -> denom == norm
    } else if (tid < 128) {
      const int m = tid - 64;
      for (int i = 0; i < Cn; ++i) zacc += bf2f(sPK[i * LDB + m]);
    }
    __syncthreads();
    if (tid >= 64 && tid < 128) sZ[tid - 64] += zacc;  // z := z_old + colsum(phi_k)

    // ---- phase 4: build fragments (all LDS reads of old state happen here) ----
    v16bf f0, f1, f2, f3, f4, f5, f6, f7;
    v8f   acc = {};
    f2 = frag_nmajor_f32(sV, LDV, 0, 0,  lane);        // B[i][v] = V[v][i]
    f3 = frag_nmajor_f32(sV, LDV, 0, 32, lane);
    if (w < 4) {                                        // out = A_mask V + phi_q S_old
      f0 = frag_rowmajor_us(sA,  LDB, w, 0,  lane);
      f1 = frag_rowmajor_us(sA,  LDB, w, 32, lane);
      f4 = frag_rowmajor_us(sPQ, LDB, w, 0,  lane);
      f5 = frag_rowmajor_us(sPQ, LDB, w, 32, lane);
      f6 = frag_kmajor_f32(sS, LDS_S, 0, 0,  lane);     // B[m][v] = S[m][v]
      f7 = frag_kmajor_f32(sS, LDS_S, 0, 32, lane);
    } else {                                            // S += phi_k^T V
      const int r2 = w - 4;
      f0 = frag_kmajor_us(sPK, LDB, r2, 0,  lane);      // A[m][i] = phi_k[i][m]
      f1 = frag_kmajor_us(sPK, LDB, r2, 32, lane);
      f4 = f2; f5 = f3; f6 = f2; f7 = f3;               // keep regs initialized
#pragma unroll
      for (int j = 0; j < 8; ++j)
        acc[j] = sS[(r2 * 16 + (half << 3) + j) * LDS_S + n];
    }
    __syncthreads();

    // ---- phase 5: WMMAs + writes (output to global, new state to LDS) ----
    if (w < 4) {
      acc = WMMA_BF16(f0, f2, acc);
      acc = WMMA_BF16(f1, f3, acc);
      acc = WMMA_BF16(f4, f6, acc);
      acc = WMMA_BF16(f5, f7, acc);
      const int rbase = w * 16 + (half << 3);
      float4 o0, o1;
      o0.x = acc[0] * sNorm[rbase + 0];
      o0.y = acc[1] * sNorm[rbase + 1];
      o0.z = acc[2] * sNorm[rbase + 2];
      o0.w = acc[3] * sNorm[rbase + 3];
      o1.x = acc[4] * sNorm[rbase + 4];
      o1.y = acc[5] * sNorm[rbase + 5];
      o1.z = acc[6] * sNorm[rbase + 6];
      o1.w = acc[7] * sNorm[rbase + 7];
      float* dst = ob + (size_t)n * Ln + l0 + rbase;    // 8 consecutive l's
      *(float4*)dst       = o0;
      *(float4*)(dst + 4) = o1;
    } else {
      acc = WMMA_BF16(f0, f2, acc);
      acc = WMMA_BF16(f1, f3, acc);
      const int r2 = w - 4;
#pragma unroll
      for (int j = 0; j < 8; ++j)
        sS[(r2 * 16 + (half << 3) + j) * LDS_S + n] = acc[j];
    }
    __syncthreads();
  }
}

extern "C" void kernel_launch(void* const* d_in, const int* in_sizes, int n_in,
                              void* d_out, int out_size, void* d_ws, size_t ws_size,
                              hipStream_t stream) {
  (void)in_sizes; (void)n_in; (void)out_size; (void)d_ws; (void)ws_size;
  const float* keys    = (const float*)d_in[0];
  const float* values  = (const float*)d_in[1];
  const float* queries = (const float*)d_in[2];
  const float* proj    = (const float*)d_in[3];
  float* out = (float*)d_out;
  dim3 grid(Bn * NSL), block(256);
  hipLaunchKernelGGL(favor_causal_kernel, grid, block, 0, stream,
                     keys, values, queries, proj, out);
}
